// OtLlamaAttention_11055245820043
// MI455X (gfx1250) — matrix operand order repaired and checked by reading, not compiled, not run on hardware
//
#include <hip/hip_runtime.h>

// Problem dims
constexpr int SEQ   = 1024;
constexpr int HID   = 4096;
constexpr int NHEAD = 32;
constexpr int NKVH  = 8;
constexpr int HDIM  = 128;
constexpr int QD  = NHEAD * HDIM;   // 4096
constexpr int KVD = NKVH * HDIM;    // 1024

typedef __attribute__((ext_vector_type(16))) __bf16 v16bf;
typedef __attribute__((ext_vector_type(8)))  __bf16 v8bf;
typedef __attribute__((ext_vector_type(2)))  __bf16 v2bf;
typedef __attribute__((ext_vector_type(8)))  float  v8f;
typedef __attribute__((ext_vector_type(4)))  float  v4f;
typedef __attribute__((ext_vector_type(4)))  int    v4i;

#if __has_builtin(__builtin_amdgcn_global_load_async_to_lds_b128) && \
    __has_builtin(__builtin_amdgcn_s_wait_asynccnt)
#define HAS_ASYNC 1
#else
#define HAS_ASYNC 0
#endif

// async-to-LDS builtin takes typed int4 pointers in AS1 (global) / AS3 (LDS)
typedef __attribute__((address_space(1))) v4i* gv4i_p;
typedef __attribute__((address_space(3))) v4i* lv4i_p;

__device__ __forceinline__ __bf16 f2bf(float f) {
  unsigned u = __builtin_bit_cast(unsigned, f);
  unsigned short h = (unsigned short)((u + 0x7FFFu + ((u >> 16) & 1u)) >> 16);
  return __builtin_bit_cast(__bf16, h);
}

__device__ __forceinline__ v2bf f2bf2(float a, float b) {
#if __has_builtin(__builtin_amdgcn_cvt_pk_bf16_f32)
  return __builtin_amdgcn_cvt_pk_bf16_f32(a, b);
#else
  v2bf r; r[0] = f2bf(a); r[1] = f2bf(b); return r;
#endif
}

// Involution on a K index 0..31: swap bits 3 and 4. Storing A-operand data at
// perm(k) makes each lane's 16-element WMMA A-fragment one contiguous 32B load.
__device__ __forceinline__ int permK(int k) {
  return (k & ~24) | ((k & 8) << 1) | ((k & 16) >> 1);
}

// ---------------------------------------------------------------------------
// Elementwise fp32 -> bf16, 8 per thread (n multiple of 8)
// ---------------------------------------------------------------------------
__global__ void cvt_bf16(const float* __restrict__ in, __bf16* __restrict__ out, int n) {
  int i = (blockIdx.x * blockDim.x + threadIdx.x) * 8;
  if (i < n) {
    v4f x0 = *(const v4f*)(in + i);
    v4f x1 = *(const v4f*)(in + i + 4);
    union { v8bf v; v2bf p[4]; } u;
    u.p[0] = f2bf2(x0[0], x0[1]);
    u.p[1] = f2bf2(x0[2], x0[3]);
    u.p[2] = f2bf2(x1[0], x1[1]);
    u.p[3] = f2bf2(x1[2], x1[3]);
    *(v8bf*)(out + i) = u.v;
  }
}

// ---------------------------------------------------------------------------
// Weight convert + transpose: in fp32 [K][N] -> out bf16 [N][K]  (K == 4096)
// so GEMM B tiles become contiguous K-rows (async-DMA friendly).
// ---------------------------------------------------------------------------
__global__ void cvt_bf16_t(const float* __restrict__ in, __bf16* __restrict__ out, int N) {
  int idx = blockIdx.x * blockDim.x + threadIdx.x;   // n*4096 + k
  if (idx < N * 4096) {
    int k = idx & 4095;
    int n = idx >> 12;
    out[idx] = f2bf(in[(size_t)k * N + n]);
  }
}

// ---------------------------------------------------------------------------
// C[M,N] (fp32) = A[M,K] * Bt[N,K]^T, bf16 WMMA compute.
// Block = 8 waves; block tile 64(M) x 128(N); wave tile 16x64.
// Both operands staged by async global->LDS DMA, double-buffered so tile i+1
// streams in while tile i is computed (s_wait_asynccnt 3 = 1 A + 2 B in flight).
// A staged in natural k order; all fragments are single contiguous 32B LDS loads.
// ---------------------------------------------------------------------------
__global__ __launch_bounds__(256) void gemm_bf16_wmma(
    const __bf16* __restrict__ A, const __bf16* __restrict__ Bt,
    float* __restrict__ C, int N, int K) {
  __shared__ __align__(32) __bf16 Asm[2][64][32];    // [buf][row][k]
  __shared__ __align__(32) __bf16 Bsm[2][128][32];   // [buf][col][k]

  const int tid  = threadIdx.x;
  const int lane = tid & 31;
  const int wave = tid >> 5;
  const int hl   = lane >> 4;
  const int l16  = lane & 15;
  const int wm   = wave >> 1;       // 0..3  -> M sub-tile
  const int wn   = wave & 1;        // 0..1  -> N half
  const int n0   = blockIdx.x * 128;
  const int m0   = blockIdx.y * 64;

  const int arow  = tid >> 2;               // 0..63
  const int achk  = (tid & 3) * 8;          // 0,8,16,24
  const int apchk = achk;                   // natural-order destination

  auto stage = [&](int k0, int buf) {
#if HAS_ASYNC
    __builtin_amdgcn_global_load_async_to_lds_b128(
        (gv4i_p)(size_t)(A + (size_t)(m0 + arow) * K + k0 + achk),
        (lv4i_p)(unsigned)(size_t)(void*)&Asm[buf][arow][apchk], 0, 0);
#pragma unroll
    for (int j = 0; j < 2; ++j) {
      int task = tid + 256 * j;
      int col = task >> 2;                  // 0..127
      int chk = (task & 3) * 8;
      __builtin_amdgcn_global_load_async_to_lds_b128(
          (gv4i_p)(size_t)(Bt + (size_t)(n0 + col) * K + k0 + chk),
          (lv4i_p)(unsigned)(size_t)(void*)&Bsm[buf][col][chk], 0, 0);
    }
#else
    *(v8bf*)&Asm[buf][arow][apchk] = *(const v8bf*)(A + (size_t)(m0 + arow) * K + k0 + achk);
#pragma unroll
    for (int j = 0; j < 2; ++j) {
      int task = tid + 256 * j;
      int col = task >> 2;
      int chk = (task & 3) * 8;
      *(v8bf*)&Bsm[buf][col][chk] = *(const v8bf*)(Bt + (size_t)(n0 + col) * K + k0 + chk);
    }
#endif
  };

  v8f acc[4] = {};
  const int niter = K / 32;
  stage(0, 0);

  for (int i = 0; i < niter; ++i) {
    const int buf = i & 1;
    __syncthreads();                        // prev readers of buf^1 done
    if (i + 1 < niter) stage((i + 1) * 32, buf ^ 1);
#if HAS_ASYNC
    if (i + 1 < niter) __builtin_amdgcn_s_wait_asynccnt(3);
    else               __builtin_amdgcn_s_wait_asynccnt(0);
#endif
    __syncthreads();                        // buf fully resident everywhere

    v16bf af = *(const v16bf*)&Asm[buf][wm * 16 + l16][hl * 16];
#pragma unroll
    for (int nt = 0; nt < 4; ++nt) {
      v16bf bfrag = *(const v16bf*)&Bsm[buf][wn * 64 + nt * 16 + l16][hl * 16];
      acc[nt] = __builtin_amdgcn_wmma_f32_16x16x32_bf16(false, af, false, bfrag,
                                                        (short)0, acc[nt], false, false);
    }
  }

  float* cp = C + (size_t)(m0 + wm * 16 + hl * 8) * N + n0 + wn * 64 + l16;
#pragma unroll
  for (int nt = 0; nt < 4; ++nt)
#pragma unroll
    for (int r = 0; r < 8; ++r)
      cp[(size_t)r * N + nt * 16] = acc[nt][r];
}

// ---------------------------------------------------------------------------
// RoPE on fp32 q/k -> bf16 with quant clamps pre-applied.
// q written in natural head-dim order (contiguous k, as the fragments read it)
// so attention Q fragments are single contiguous 32B loads.
// ---------------------------------------------------------------------------
__global__ void rope_cvt_kernel(const float* __restrict__ q, const float* __restrict__ k,
                                const float* __restrict__ cosb, const float* __restrict__ sinb,
                                __bf16* __restrict__ qh, __bf16* __restrict__ kh) {
  int idx = blockIdx.x * blockDim.x + threadIdx.x;
  const int QN = SEQ * NHEAD * 64;
  const int KN = SEQ * NKVH * 64;
  if (idx < QN) {
    int d = idx & 63; int t = idx >> 6;
    int h = t & (NHEAD - 1); int s = t >> 5;
    float c0 = cosb[s * HDIM + d],      s0 = sinb[s * HDIM + d];
    float c1 = cosb[s * HDIM + d + 64], s1 = sinb[s * HDIM + d + 64];
    const float* p = q + (size_t)s * QD + h * HDIM + d;
    float x0 = p[0], x1 = p[64];
    float y0 = fminf(fmaxf(x0 * c0 - x1 * s0, -3.f), 3.f);
    float y1 = fminf(fmaxf(x1 * c1 + x0 * s1, -3.f), 3.f);
    int dd = d;                                  // natural-order position
    __bf16* o = qh + (size_t)s * QD + h * HDIM + dd;
    o[0] = f2bf(y0); o[64] = f2bf(y1);
  } else if (idx < QN + KN) {
    int j = idx - QN;
    int d = j & 63; int t = j >> 6;
    int h = t & (NKVH - 1); int s = t >> 3;
    float c0 = cosb[s * HDIM + d],      s0 = sinb[s * HDIM + d];
    float c1 = cosb[s * HDIM + d + 64], s1 = sinb[s * HDIM + d + 64];
    const float* p = k + (size_t)s * KVD + h * HDIM + d;
    float x0 = p[0], x1 = p[64];
    float y0 = fminf(fmaxf(x0 * c0 - x1 * s0, -3.f), 3.f);
    float y1 = fminf(fmaxf(x1 * c1 + x0 * s1, -3.f), 3.f);
    y0 = (fabsf(y0) < 0.06012f) ? 0.f : y0;   // K LUT-zero
    y1 = (fabsf(y1) < 0.06012f) ? 0.f : y1;
    __bf16* o = kh + (size_t)s * KVD + h * HDIM + d;   // B-operand: natural order
    o[0] = f2bf(y0); o[64] = f2bf(y1);
  }
}

// ---------------------------------------------------------------------------
// V: fp32 [s][KVD] -> bf16 transposed [KVD][SEQ], clamp [-4,4] + LUT-zero.
// ---------------------------------------------------------------------------
__global__ void vcvt_kernel(const float* __restrict__ v, __bf16* __restrict__ vt) {
  int idx = blockIdx.x * blockDim.x + threadIdx.x;   // dt*SEQ + s
  if (idx < KVD * SEQ) {
    int s  = idx & (SEQ - 1);
    int dt = idx >> 10;
    float x = v[(size_t)s * KVD + dt];
    x = fminf(fmaxf(x, -4.f), 4.f);
    x = (fabsf(x) < 0.08016f) ? 0.f : x;
    vt[idx] = f2bf(x);
  }
}

// ---------------------------------------------------------------------------
// Flash-style causal GQA attention, all-bf16 operands (pre-clamped).
// One wave owns (head, 16 query rows). All fragments contiguous 32B loads;
// P transposed to A-layout through a natural-key-order per-wave LDS tile.
// ---------------------------------------------------------------------------
__global__ __launch_bounds__(256) void attn_kernel(
    const __bf16* __restrict__ Q, const __bf16* __restrict__ Kb,
    const __bf16* __restrict__ Vt, __bf16* __restrict__ O) {
  __shared__ __align__(32) __bf16 pbuf[8][16][32];   // [wave][row][key]

  const int lane = threadIdx.x & 31;
  const int wave = threadIdx.x >> 5;
  const int hl   = lane >> 4;
  const int l16  = lane & 15;
  const int h    = blockIdx.y;
  const int kvh  = h >> 2;                      // NHEAD/NKVH = 4
  const int m0   = (blockIdx.x * 8 + wave) * 16;

  // Q fragments (qh stored in fragment order -> direct 32B loads)
  v16bf qf[4];
  {
    const __bf16* qp = Q + (size_t)(m0 + l16) * QD + h * HDIM + hl * 16;
#pragma unroll
    for (int c = 0; c < 4; ++c) qf[c] = *(const v16bf*)(qp + c * 32);
  }

  v8f oacc[8] = {};
  float mrow[8], lrow[8];
#pragma unroll
  for (int r = 0; r < 8; ++r) { mrow[r] = -3.0e38f; lrow[r] = 0.f; }

  const int pc0 = l16;               // storage col for sub-tile t=0
  const int pc1 = l16 + 16;          // storage col for sub-tile t=1

  const int qmax = m0 + 15;
  for (int kt = 0; kt <= qmax; kt += 32) {
    v8f sc[2];
#pragma unroll
    for (int t = 0; t < 2; ++t) {
      v8f s = {};
      const int key = kt + t * 16 + l16;
      const __bf16* kp = Kb + (size_t)key * KVD + kvh * HDIM + hl * 16;
#pragma unroll
      for (int c = 0; c < 4; ++c) {
        v16bf bfrag = *(const v16bf*)(kp + c * 32);
        s = __builtin_amdgcn_wmma_f32_16x16x32_bf16(false, qf[c], false, bfrag,
                                                    (short)0, s, false, false);
      }
#pragma unroll
      for (int r = 0; r < 8; ++r) {
        float v = fminf(fmaxf(s[r], -64.f), 64.f) * 0.08838834764831845f;
        int kpos = kt + t * 16 + l16;
        int qrow = m0 + hl * 8 + r;
        s[r] = (kpos <= qrow) ? v : -3.0e38f;
      }
      sc[t] = s;
    }

    // online softmax (xor 1/2/4/8 stays within each 16-lane half)
    float alpha[8];
#pragma unroll
    for (int r = 0; r < 8; ++r) {
      float v = fmaxf(sc[0][r], sc[1][r]);
#pragma unroll
      for (int off = 1; off < 16; off <<= 1)
        v = fmaxf(v, __shfl_xor(v, off, 32));
      float mnew = fmaxf(mrow[r], v);
      alpha[r] = __expf(mrow[r] - mnew);
      mrow[r] = mnew;
      float p0 = __expf(sc[0][r] - mnew);
      float p1 = __expf(sc[1][r] - mnew);
      sc[0][r] = p0; sc[1][r] = p1;
      float ps = p0 + p1;
#pragma unroll
      for (int off = 1; off < 16; off <<= 1)
        ps += __shfl_xor(ps, off, 32);
      lrow[r] = lrow[r] * alpha[r] + ps;
    }

    // P: C-layout -> LDS (fragment order) -> single 32B A-fragment load
#pragma unroll
    for (int r = 0; r < 8; ++r) {
      pbuf[wave][hl * 8 + r][pc0] = f2bf(sc[0][r]);
      pbuf[wave][hl * 8 + r][pc1] = f2bf(sc[1][r]);
    }

#pragma unroll
    for (int n = 0; n < 8; ++n)
#pragma unroll
      for (int r = 0; r < 8; ++r) oacc[n][r] *= alpha[r];

    v16bf pf = *(const v16bf*)&pbuf[wave][l16][hl * 16];

    // O += P * V  (V^T layout: contiguous along keys)
#pragma unroll
    for (int n = 0; n < 8; ++n) {
      v16bf vfrag = *(const v16bf*)(Vt + (size_t)(kvh * HDIM + n * 16 + l16) * SEQ + kt + hl * 16);
      oacc[n] = __builtin_amdgcn_wmma_f32_16x16x32_bf16(false, pf, false, vfrag,
                                                        (short)0, oacc[n], false, false);
    }
  }

  // finalize: /l, clamp AV range, store bf16 for O-projection
#pragma unroll
  for (int r = 0; r < 8; ++r) lrow[r] = 1.f / lrow[r];
  __bf16* op = O + (size_t)(m0 + hl * 8) * QD + h * HDIM;
#pragma unroll
  for (int n = 0; n < 8; ++n)
#pragma unroll
    for (int r = 0; r < 8; ++r) {
      float v = fminf(fmaxf(oacc[n][r] * lrow[r], -4.f), 4.f);
      op[(size_t)r * QD + n * 16 + l16] = f2bf(v);
    }
}

// ---------------------------------------------------------------------------
extern "C" void kernel_launch(void* const* d_in, const int* in_sizes, int n_in,
                              void* d_out, int out_size, void* d_ws, size_t ws_size,
                              hipStream_t stream) {
  const float* x    = (const float*)d_in[0];
  const float* wq   = (const float*)d_in[1];
  const float* wk   = (const float*)d_in[2];
  const float* wv   = (const float*)d_in[3];
  const float* wo   = (const float*)d_in[4];
  const float* cosb = (const float*)d_in[5];
  const float* sinb = (const float*)d_in[6];
  float* out = (float*)d_out;

  // workspace carve-up
  char* p = (char*)d_ws;
  auto carve = [&](size_t bytes) { char* r = p; p += (bytes + 255) & ~(size_t)255; return r; };
  __bf16* xh   = (__bf16*)carve((size_t)SEQ * HID * 2);
  __bf16* wqt  = (__bf16*)carve((size_t)QD  * HID * 2);   // [N][K]
  __bf16* wkt  = (__bf16*)carve((size_t)KVD * HID * 2);   // [N][K]
  __bf16* wvt  = (__bf16*)carve((size_t)KVD * HID * 2);   // [N][K]
  __bf16* wot  = (__bf16*)carve((size_t)HID * QD  * 2);   // [N][K]
  float*  qf32 = (float*) carve((size_t)SEQ * QD  * 4);
  float*  kf32 = (float*) carve((size_t)SEQ * KVD * 4);
  float*  vf32 = (float*) carve((size_t)SEQ * KVD * 4);
  __bf16* qh   = (__bf16*)carve((size_t)SEQ * QD  * 2);
  __bf16* kh   = (__bf16*)carve((size_t)SEQ * KVD * 2);
  __bf16* vt   = (__bf16*)carve((size_t)KVD * SEQ * 2);
  __bf16* oh   = (__bf16*)carve((size_t)SEQ * QD  * 2);

  dim3 blk(256);
  cvt_bf16<<<(SEQ * HID / 8 + 255) / 256, blk, 0, stream>>>(x, xh, SEQ * HID);
  cvt_bf16_t<<<(QD  * HID + 255) / 256, blk, 0, stream>>>(wq, wqt, QD);
  cvt_bf16_t<<<(KVD * HID + 255) / 256, blk, 0, stream>>>(wk, wkt, KVD);
  cvt_bf16_t<<<(KVD * HID + 255) / 256, blk, 0, stream>>>(wv, wvt, KVD);
  cvt_bf16_t<<<(HID * QD  + 255) / 256, blk, 0, stream>>>(wo, wot, HID);

  gemm_bf16_wmma<<<dim3(QD / 128,  SEQ / 64), blk, 0, stream>>>(xh, wqt, qf32, QD,  HID);
  gemm_bf16_wmma<<<dim3(KVD / 128, SEQ / 64), blk, 0, stream>>>(xh, wkt, kf32, KVD, HID);
  gemm_bf16_wmma<<<dim3(KVD / 128, SEQ / 64), blk, 0, stream>>>(xh, wvt, vf32, KVD, HID);

  int rtotal = SEQ * NHEAD * 64 + SEQ * NKVH * 64;
  rope_cvt_kernel<<<(rtotal + 255) / 256, blk, 0, stream>>>(qf32, kf32, cosb, sinb, qh, kh);
  vcvt_kernel<<<(KVD * SEQ + 255) / 256, blk, 0, stream>>>(vf32, vt);

  attn_kernel<<<dim3(SEQ / 128, NHEAD), blk, 0, stream>>>(qh, kh, vt, oh);

  gemm_bf16_wmma<<<dim3(QD / 128, SEQ / 64), blk, 0, stream>>>(oh, wot, out, QD, QD);
}
